// ProofOptimizationGNN_68470368633094
// MI455X (gfx1250) — compile-verified, matrix operand order checked by reading, not run on hardware
//
#include <hip/hip_runtime.h>
#include <hip/hip_bf16.h>

// ---------------------------------------------------------------------------
// Problem constants (match reference)
// ---------------------------------------------------------------------------
#define GN 100000   // nodes
#define GE 600000   // edges
#define GF 64       // node feature dim
#define GED 16      // edge feature dim
#define GH 128      // hidden
#define GL 4        // layers
#define GO 32       // outputs

typedef __attribute__((ext_vector_type(16))) __bf16 v16bf;
typedef __attribute__((ext_vector_type(8)))  __bf16 v8bf;
typedef __attribute__((ext_vector_type(8)))  float  v8f;

// ---------------------------------------------------------------------------
// WMMA fragment helpers (CDNA5 wave32 layouts, ISA 7.12.2)
//   A 16x32 bf16: lane L holds row m=L%16.
//     lanes 0-15 : elems 0..7 -> K = k0+0..7,   elems 8..15 -> K = k0+16..23
//     lanes 16-31: elems 0..7 -> K = k0+8..15,  elems 8..15 -> K = k0+24..31
//   B 32x16 bf16: same K grouping, lane L holds column n=L%16.
//   C/D 16x16 f32: reg r, lane L -> M = r + 8*(L>=16), N = L%16.
// ---------------------------------------------------------------------------
__device__ __forceinline__ v16bf load_a_lds(const __bf16* A, int lda, int k0, int lane) {
    const __bf16* p = A + (lane & 15) * lda + k0 + ((lane & 16) ? 8 : 0);
    v8bf lo = *(const v8bf*)p;          // K = g0 .. g0+7   (16B aligned)
    v8bf hi = *(const v8bf*)(p + 16);   // K = g0+16 .. g0+23
    return __builtin_shufflevector(lo, hi, 0, 1, 2, 3, 4, 5, 6, 7,
                                          8, 9, 10, 11, 12, 13, 14, 15);
}

// B fragment from prepacked bf16 weights: one contiguous 32B load per lane.
// Packed layout: [colTile n0][kChunk kc][lane 0..31][elem 0..15]
__device__ __forceinline__ v16bf load_b_pk(const __bf16* P, int nkc, int kc,
                                           int wv, int lane) {
    return ((const v16bf*)P)[((size_t)(wv * nkc + kc) * 32) + lane];
}

__device__ __forceinline__ v8f wmma_bf16(v16bf a, v16bf b, v8f c) {
    return __builtin_amdgcn_wmma_f32_16x16x32_bf16(false, a, false, b,
                                                   (short)0, c, false, false);
}

// Async global->LDS 16-byte copy (CDNA5 TDM-class async path, ASYNCcnt).
// GV mode: vdst = LDS byte address, vaddr = 64-bit global address.
__device__ __forceinline__ void async_cp16(void* lds_ptr, const void* gptr) {
    unsigned lds = (unsigned)(uintptr_t)lds_ptr;   // flat->LDS offset (addr[31:0])
    asm volatile("global_load_async_to_lds_b128 %0, %1, off"
                 :: "v"(lds), "v"(gptr) : "memory");
}
__device__ __forceinline__ void async_wait0() {
    asm volatile("s_wait_asynccnt 0x0" ::: "memory");
}

// 8 consecutive f32 -> v8bf (optionally scaled)
__device__ __forceinline__ v8bf cvt8(const float* p, float s) {
    float4 a = *(const float4*)p;
    float4 b = *(const float4*)(p + 4);
    v8bf v;
    v[0] = (__bf16)(a.x * s); v[1] = (__bf16)(a.y * s);
    v[2] = (__bf16)(a.z * s); v[3] = (__bf16)(a.w * s);
    v[4] = (__bf16)(b.x * s); v[5] = (__bf16)(b.y * s);
    v[6] = (__bf16)(b.z * s); v[7] = (__bf16)(b.w * s);
    return v;
}

// 16-lane group sum (lanes [0..15] / [16..31] reduce independently, wave32)
__device__ __forceinline__ float grp16_sum(float s) {
    s += __shfl_xor(s, 1);
    s += __shfl_xor(s, 2);
    s += __shfl_xor(s, 4);
    s += __shfl_xor(s, 8);
    return s;
}

// ---------------------------------------------------------------------------
// Weight prepack: f32 row-major W[kmax x (ncolt*16)] -> bf16 B-fragment layout,
// K zero-padded to nkc*32. Runs once per launch; output lives in L2.
// ---------------------------------------------------------------------------
__global__ void prepack_kernel(const float* __restrict__ W, __bf16* __restrict__ P,
                               int kmax, int nkc, int ncolt, int ldw) {
    int total = ncolt * nkc * 512;
    for (int idx = blockIdx.x * 256 + threadIdx.x; idx < total;
         idx += gridDim.x * 256) {
        int elem = idx & 15;
        int lane = (idx >> 4) & 31;
        int tmp  = idx >> 9;
        int kc   = tmp % nkc;
        int n0   = tmp / nkc;
        int g0   = kc * 32 + ((lane & 16) ? 8 : 0);
        int ka   = (elem < 8) ? (g0 + elem) : (g0 + 8 + elem);  // +16+(elem-8)
        int n    = n0 * 16 + (lane & 15);
        P[idx]   = (ka < kmax) ? (__bf16)W[(size_t)ka * ldw + n] : (__bf16)0.f;
    }
}

// ---------------------------------------------------------------------------
// in-degree counts (float, matches segment_sum of ones)
// ---------------------------------------------------------------------------
__global__ void degree_kernel(const long long* __restrict__ ei, float* counts) {
    long e = (long)blockIdx.x * 256 + threadIdx.x;
    if (e < GE) atomicAdd(&counts[(int)ei[GE + e]], 1.0f);
}

// ---------------------------------------------------------------------------
// Node encoder: relu(x@W1+b1) -> LN -> relu(@W2+b2). 16 nodes / block.
// ---------------------------------------------------------------------------
__global__ void encoder_kernel(const float* __restrict__ x,
                               const __bf16* __restrict__ pw1, const float* __restrict__ b1,
                               const float* __restrict__ lng, const float* __restrict__ lnb,
                               const __bf16* __restrict__ pw2, const float* __restrict__ b2,
                               float* __restrict__ hf, __bf16* __restrict__ hb) {
    __shared__ __bf16 sA[16 * GF];       // node features, bf16
    __shared__ float  sT[16 * GH];       // relu(x@W1+b1)
    __shared__ __bf16 sB[16 * GH];       // LN output, bf16 (A of 2nd GEMM)
    int t = threadIdx.x, lane = t & 31, wv = t >> 5;
    long row0 = (long)blockIdx.x * 16;

    for (int i = t; i < 16 * 8; i += 256) {   // 16 rows * 8 chunks of 8 floats
        int r = i >> 3, c8 = i & 7;
        *(v8bf*)(sA + r * GF + c8 * 8) = cvt8(x + (row0 + r) * GF + c8 * 8, 1.f);
    }
    __syncthreads();

    int n = wv * 16 + (lane & 15), mb = (lane & 16) ? 8 : 0;
    v8f acc = {};
#pragma unroll
    for (int kc = 0; kc < 2; ++kc)
        acc = wmma_bf16(load_a_lds(sA, GF, kc * 32, lane),
                        load_b_pk(pw1, 2, kc, wv, lane), acc);
#pragma unroll
    for (int r = 0; r < 8; ++r) {
        float v = acc[r] + b1[n];
        sT[(mb + r) * GH + n] = v > 0.f ? v : 0.f;
    }
    __syncthreads();

    {   // parallel LayerNorm: 16 threads per node, 8 elems each
        int r = t >> 4, j = t & 15;
        const float* pr = sT + r * GH + j * 8;
        float4 a = *(const float4*)pr, b = *(const float4*)(pr + 4);
        float vals[8] = {a.x, a.y, a.z, a.w, b.x, b.y, b.z, b.w};
        float s = vals[0] + vals[1] + vals[2] + vals[3] +
                  vals[4] + vals[5] + vals[6] + vals[7];
        float mu = grp16_sum(s) * (1.f / GH);
        float vv = 0.f;
#pragma unroll
        for (int k = 0; k < 8; ++k) { float d = vals[k] - mu; vv += d * d; }
        float inv = rsqrtf(grp16_sum(vv) * (1.f / GH) + 1e-5f);
        v8bf o;
#pragma unroll
        for (int k = 0; k < 8; ++k)
            o[k] = (__bf16)((vals[k] - mu) * inv * lng[j * 8 + k] + lnb[j * 8 + k]);
        *(v8bf*)(sB + r * GH + j * 8) = o;
    }
    __syncthreads();

    v8f acc2 = {};
#pragma unroll
    for (int kc = 0; kc < 4; ++kc)
        acc2 = wmma_bf16(load_a_lds(sB, GH, kc * 32, lane),
                         load_b_pk(pw2, 4, kc, wv, lane), acc2);
#pragma unroll
    for (int r = 0; r < 8; ++r) {
        float v = acc2[r] + b2[n];
        v = v > 0.f ? v : 0.f;
        size_t idx = (size_t)(row0 + mb + r) * GH + n;
        hf[idx] = v;
        hb[idx] = (__bf16)v;
    }
}

// ---------------------------------------------------------------------------
// Message MLP over a 16-edge tile + atomic scatter-add into agg.
//   m_in = [h_bf16[src] | h_bf16[dst] | ef] : 272 wide, padded to 288 (9 K-steps)
// Gather chunks go through the CDNA5 async global->LDS path (ASYNCcnt).
// ---------------------------------------------------------------------------
__global__ void message_kernel(const __bf16* __restrict__ hb,
                               const long long* __restrict__ ei,
                               const float* __restrict__ ef,
                               const __bf16* __restrict__ pw1, const float* __restrict__ b1,
                               const __bf16* __restrict__ pw2, const float* __restrict__ b2,
                               float* __restrict__ agg) {
    __shared__ int    sSrc[16], sDst[16];
    __shared__ __bf16 sIn[16 * 288];
    __shared__ __bf16 sH1[16 * GH];
    int t = threadIdx.x, lane = t & 31, wv = t >> 5;
    long e0 = (long)blockIdx.x * 16;

    if (t < 16) {
        sSrc[t] = (int)ei[e0 + t];
        sDst[t] = (int)ei[(long)GE + e0 + t];
    }
    __syncthreads();

    for (int i = t; i < 16 * 36; i += 256) {  // 36 chunks of 8 bf16 per row
        int r = i / 36, c8 = i - r * 36;
        __bf16* dst = sIn + r * 288 + c8 * 8;
        if (c8 < 16) {          // h[src] : async 16B global->LDS copy
            async_cp16(dst, hb + (size_t)sSrc[r] * GH + c8 * 8);
        } else if (c8 < 32) {   // h[dst]
            async_cp16(dst, hb + (size_t)sDst[r] * GH + (c8 - 16) * 8);
        } else if (c8 < 34) {   // edge features, f32 -> bf16
            *(v8bf*)dst = cvt8(ef + (e0 + r) * GED + (c8 - 32) * 8, 1.f);
        } else {                // K pad 272..287
            v8bf z;
#pragma unroll
            for (int j = 0; j < 8; ++j) z[j] = (__bf16)0.f;
            *(v8bf*)dst = z;
        }
    }
    async_wait0();              // async copies landed in LDS
    __syncthreads();

    int n = wv * 16 + (lane & 15), mb = (lane & 16) ? 8 : 0;
    v8f acc = {};
#pragma unroll
    for (int kc = 0; kc < 9; ++kc)
        acc = wmma_bf16(load_a_lds(sIn, 288, kc * 32, lane),
                        load_b_pk(pw1, 9, kc, wv, lane), acc);
#pragma unroll
    for (int r = 0; r < 8; ++r) {
        float v = acc[r] + b1[n];
        sH1[(mb + r) * GH + n] = (__bf16)(v > 0.f ? v : 0.f);
    }
    __syncthreads();

    v8f acc2 = {};
#pragma unroll
    for (int kc = 0; kc < 4; ++kc)
        acc2 = wmma_bf16(load_a_lds(sH1, GH, kc * 32, lane),
                         load_b_pk(pw2, 4, kc, wv, lane), acc2);
#pragma unroll
    for (int r = 0; r < 8; ++r)
        atomicAdd(&agg[(size_t)sDst[mb + r] * GH + n], acc2[r] + b2[n]);
}

// ---------------------------------------------------------------------------
// Update: h_new = relu([h | agg*inv_cnt] @ Wu + bu); h = LN(h + h_new)
// Also zeroes agg after reading (ready for the next layer / next launch).
// ---------------------------------------------------------------------------
__global__ void update_kernel(const float* __restrict__ counts,
                              float* __restrict__ agg,
                              const __bf16* __restrict__ puw, const float* __restrict__ ub,
                              const float* __restrict__ lng, const float* __restrict__ lnb,
                              float* __restrict__ hf, __bf16* __restrict__ hb) {
    __shared__ __bf16 sA[16 * 256];
    __shared__ float  sR[16 * GH];
    int t = threadIdx.x, lane = t & 31, wv = t >> 5;
    long row0 = (long)blockIdx.x * 16;

    for (int i = t; i < 16 * 32; i += 256) {  // 32 chunks of 8 per row
        int r = i >> 5, c8 = i & 31;
        long node = row0 + r;
        v8bf v;
        if (c8 < 16) {
            v = cvt8(hf + (size_t)node * GH + c8 * 8, 1.f);
        } else {
            float* ap = agg + (size_t)node * GH + (c8 - 16) * 8;
            v = cvt8(ap, 1.f / (counts[node] + 1e-8f));
            float4 z = make_float4(0.f, 0.f, 0.f, 0.f);
            ((float4*)ap)[0] = z;             // zero for next layer
            ((float4*)ap)[1] = z;
        }
        *(v8bf*)(sA + r * 256 + c8 * 8) = v;
    }
    __syncthreads();

    int n = wv * 16 + (lane & 15), mb = (lane & 16) ? 8 : 0;
    v8f acc = {};
#pragma unroll
    for (int kc = 0; kc < 8; ++kc)
        acc = wmma_bf16(load_a_lds(sA, 256, kc * 32, lane),
                        load_b_pk(puw, 8, kc, wv, lane), acc);
#pragma unroll
    for (int r = 0; r < 8; ++r) {
        float v = acc[r] + ub[n];
        v = v > 0.f ? v : 0.f;
        sR[(mb + r) * GH + n] = hf[(size_t)(row0 + mb + r) * GH + n] + v;  // residual
    }
    __syncthreads();

    {   // parallel LayerNorm + vectorized writeback
        int r = t >> 4, j = t & 15;
        const float* pr = sR + r * GH + j * 8;
        float4 a = *(const float4*)pr, b = *(const float4*)(pr + 4);
        float vals[8] = {a.x, a.y, a.z, a.w, b.x, b.y, b.z, b.w};
        float s = vals[0] + vals[1] + vals[2] + vals[3] +
                  vals[4] + vals[5] + vals[6] + vals[7];
        float mu = grp16_sum(s) * (1.f / GH);
        float vv = 0.f;
#pragma unroll
        for (int k = 0; k < 8; ++k) { float d = vals[k] - mu; vv += d * d; }
        float inv = rsqrtf(grp16_sum(vv) * (1.f / GH) + 1e-5f);
        float o[8];
#pragma unroll
        for (int k = 0; k < 8; ++k)
            o[k] = (vals[k] - mu) * inv * lng[j * 8 + k] + lnb[j * 8 + k];
        size_t base = (size_t)(row0 + r) * GH + j * 8;
        *(float4*)(hf + base)     = make_float4(o[0], o[1], o[2], o[3]);
        *(float4*)(hf + base + 4) = make_float4(o[4], o[5], o[6], o[7]);
        v8bf ob;
#pragma unroll
        for (int k = 0; k < 8; ++k) ob[k] = (__bf16)o[k];
        *(v8bf*)(hb + base) = ob;
    }
}

// ---------------------------------------------------------------------------
// Attention scores via WMMA: s = tanh(h@W1+b1)@w2 + b2, per 16-node tile.
// Waves 0..3 each own a 16-col tile of the 64-wide hidden.
// ---------------------------------------------------------------------------
__global__ void score_kernel(const __bf16* __restrict__ hb,
                             const __bf16* __restrict__ pw1,
                             const float* __restrict__ b1,
                             const float* __restrict__ w2, const float* __restrict__ b2,
                             float* __restrict__ scores, float* __restrict__ red) {
    __shared__ __bf16 sA[16 * GH];
    __shared__ float  sT[16 * 64];
    __shared__ float  sred[16];
    int t = threadIdx.x, lane = t & 31, wv = t >> 5;
    long row0 = (long)blockIdx.x * 16;

    {   // stage h tile: async 16B copies, one chunk per thread
        int r = t >> 4, c8 = t & 15;
        async_cp16(sA + r * GH + c8 * 8,
                   hb + (size_t)(row0 + r) * GH + c8 * 8);
    }
    async_wait0();
    __syncthreads();

    if (wv < 4) {
        int n = wv * 16 + (lane & 15), mb = (lane & 16) ? 8 : 0;
        v8f acc = {};
#pragma unroll
        for (int kc = 0; kc < 4; ++kc)
            acc = wmma_bf16(load_a_lds(sA, GH, kc * 32, lane),
                            load_b_pk(pw1, 4, kc, wv, lane), acc);
        float bn = b1[n], wn = w2[n];
#pragma unroll
        for (int r = 0; r < 8; ++r)
            sT[(mb + r) * 64 + n] = tanhf(acc[r] + bn) * wn;
    }
    __syncthreads();

    {   // per-node sum of 64: 16 threads/node, 4 elems each + shuffle reduce
        int r = t >> 4, j = t & 15;
        const float* pr = sT + r * 64 + j * 4;
        float s = pr[0] + pr[1] + pr[2] + pr[3];
        s = grp16_sum(s);
        if (j == 0) {
            float sc = s + b2[0];
            scores[row0 + r] = sc;
            sred[r] = sc;
        }
    }
    __syncthreads();
    if (t == 0) {
        float m = sred[0];
        for (int i = 1; i < 16; ++i) m = fmaxf(m, sred[i]);
        red[blockIdx.x] = m;
    }
}

__global__ void final_max_kernel(const float* __restrict__ red, int nb,
                                 float* __restrict__ scal, float* __restrict__ g) {
    __shared__ float sm[512];
    float m = -3.0e38f;
    for (int i = threadIdx.x; i < nb; i += 512) m = fmaxf(m, red[i]);
    sm[threadIdx.x] = m;
    __syncthreads();
    for (int off = 256; off > 0; off >>= 1) {
        if (threadIdx.x < off)
            sm[threadIdx.x] = fmaxf(sm[threadIdx.x], sm[threadIdx.x + off]);
        __syncthreads();
    }
    if (threadIdx.x == 0) { scal[0] = sm[0]; scal[1] = 0.f; }
    if (threadIdx.x < GH) g[threadIdx.x] = 0.f;   // zero pooled accumulator
}

// e = exp(s-max); sumexp += e; g[d] += sum_n e_n * h_n[d]  (no atomic storms)
__global__ void expsum_kernel(const float* __restrict__ hf,
                              const float* __restrict__ scores,
                              const float* __restrict__ scal,
                              float* __restrict__ sumexp, float* __restrict__ g) {
    __shared__ float se[256];
    int t = threadIdx.x;
    int n0 = blockIdx.x * 256;
    int n = n0 + t;
    float e = 0.f;
    float mx = scal[0];
    if (n < GN) e = __expf(scores[n] - mx);
    se[t] = e;
    __syncthreads();

    if (t < GH) {   // one output dim per thread, coalesced hf reads
        int lim = GN - n0; if (lim > 256) lim = 256;
        float acc = 0.f;
        for (int k = 0; k < lim; ++k)
            acc += se[k] * hf[(size_t)(n0 + k) * GH + t];
        atomicAdd(&g[t], acc);
    }
    __syncthreads();
    for (int off = 128; off > 0; off >>= 1) {
        if (t < off) se[t] += se[t + off];
        __syncthreads();
    }
    if (t == 0) atomicAdd(sumexp, se[0]);
}

__global__ void head_kernel(const float* __restrict__ g, const float* __restrict__ sumexp,
                            const float* __restrict__ w1, const float* __restrict__ b1,
                            const float* __restrict__ w2, const float* __restrict__ b2,
                            float* __restrict__ out) {
    __shared__ float sgf[GH], shid[GH];
    int t = threadIdx.x;
    float inv = 1.f / sumexp[0];
    sgf[t] = g[t] * inv;
    __syncthreads();
    float a = b1[t];
    for (int d = 0; d < GH; ++d) a += sgf[d] * w1[d * GH + t];
    shid[t] = a > 0.f ? a : 0.f;
    __syncthreads();
    if (t < GO) {
        float o = b2[t];
        for (int j = 0; j < GH; ++j) o += shid[j] * w2[j * GO + t];
        out[t] = o;
    }
}

// ---------------------------------------------------------------------------
// Launch
// ---------------------------------------------------------------------------
extern "C" void kernel_launch(void* const* d_in, const int* in_sizes, int n_in,
                              void* d_out, int out_size, void* d_ws, size_t ws_size,
                              hipStream_t stream) {
    const float*     x      = (const float*)d_in[0];
    const long long* ei     = (const long long*)d_in[1];
    const float*     ef     = (const float*)d_in[2];
    const float*     enc_w1 = (const float*)d_in[3];
    const float*     enc_b1 = (const float*)d_in[4];
    const float*     enc_lg = (const float*)d_in[5];
    const float*     enc_lb = (const float*)d_in[6];
    const float*     enc_w2 = (const float*)d_in[7];
    const float*     enc_b2 = (const float*)d_in[8];
    const float*     msg_w1 = (const float*)d_in[9];
    const float*     msg_b1 = (const float*)d_in[10];
    const float*     msg_w2 = (const float*)d_in[11];
    const float*     msg_b2 = (const float*)d_in[12];
    const float*     upd_w  = (const float*)d_in[13];
    const float*     upd_b  = (const float*)d_in[14];
    const float*     ln_g   = (const float*)d_in[15];
    const float*     ln_b   = (const float*)d_in[16];
    const float*     att_w1 = (const float*)d_in[17];
    const float*     att_b1 = (const float*)d_in[18];
    const float*     att_w2 = (const float*)d_in[19];
    const float*     att_b2 = (const float*)d_in[20];
    const float*     hd_w1  = (const float*)d_in[21];
    const float*     hd_b1  = (const float*)d_in[22];
    const float*     hd_w2  = (const float*)d_in[23];
    const float*     hd_b2  = (const float*)d_in[24];
    float* out = (float*)d_out;

    // workspace carve-up (256B aligned)
    char*  ws  = (char*)d_ws;
    size_t off = 0;
    auto carve = [&](size_t bytes) {
        void* p = ws + off;
        off = (off + bytes + 255) & ~(size_t)255;
        return p;
    };
    float*  counts = (float*)carve((size_t)GN * 4);
    float*  scores = (float*)carve((size_t)GN * 4);
    float*  red    = (float*)carve(8192 * 4);    // 6250 block maxima
    float*  scal   = (float*)carve(16 * 4);      // [0]=max, [1]=sumexp
    float*  g      = (float*)carve(GH * 4);
    float*  hf     = (float*)carve((size_t)GN * GH * 4);
    __bf16* hbuf   = (__bf16*)carve((size_t)GN * GH * 2);
    float*  agg    = (float*)carve((size_t)GN * GH * 4);
    // prepacked bf16 weights (B-fragment layout): ncolt * nkc * 512 elems each
    __bf16* pk_e1  = (__bf16*)carve((size_t)8 * 2 * 512 * 2);
    __bf16* pk_e2  = (__bf16*)carve((size_t)8 * 4 * 512 * 2);
    __bf16* pk_m1  = (__bf16*)carve((size_t)GL * 8 * 9 * 512 * 2);
    __bf16* pk_m2  = (__bf16*)carve((size_t)GL * 8 * 4 * 512 * 2);
    __bf16* pk_u   = (__bf16*)carve((size_t)GL * 8 * 8 * 512 * 2);
    __bf16* pk_a1  = (__bf16*)carve((size_t)4 * 4 * 512 * 2);

    // --- weight prepack (cheap, once per launch, deterministic) ---
    prepack_kernel<<<32, 256, 0, stream>>>(enc_w1, pk_e1, GF, 2, 8, GH);
    prepack_kernel<<<64, 256, 0, stream>>>(enc_w2, pk_e2, GH, 4, 8, GH);
    for (int i = 0; i < GL; ++i) {
        prepack_kernel<<<144, 256, 0, stream>>>(msg_w1 + (size_t)i * 272 * GH,
                                                pk_m1 + (size_t)i * 8 * 9 * 512, 272, 9, 8, GH);
        prepack_kernel<<<64, 256, 0, stream>>>(msg_w2 + (size_t)i * GH * GH,
                                               pk_m2 + (size_t)i * 8 * 4 * 512, GH, 4, 8, GH);
        prepack_kernel<<<128, 256, 0, stream>>>(upd_w + (size_t)i * 256 * GH,
                                                pk_u + (size_t)i * 8 * 8 * 512, 256, 8, 8, GH);
    }
    prepack_kernel<<<32, 256, 0, stream>>>(att_w1, pk_a1, GH, 4, 4, GH / 2);

    hipMemsetAsync(counts, 0, (size_t)GN * 4, stream);
    hipMemsetAsync(agg, 0, (size_t)GN * GH * 4, stream);  // update re-zeroes each layer
    degree_kernel<<<(GE + 255) / 256, 256, 0, stream>>>(ei, counts);

    encoder_kernel<<<GN / 16, 256, 0, stream>>>(x, pk_e1, enc_b1, enc_lg, enc_lb,
                                                pk_e2, enc_b2, hf, hbuf);

    for (int i = 0; i < GL; ++i) {
        message_kernel<<<GE / 16, 256, 0, stream>>>(
            hbuf, ei, ef,
            pk_m1 + (size_t)i * 8 * 9 * 512, msg_b1 + (size_t)i * GH,
            pk_m2 + (size_t)i * 8 * 4 * 512, msg_b2 + (size_t)i * GH, agg);
        update_kernel<<<GN / 16, 256, 0, stream>>>(
            counts, agg,
            pk_u + (size_t)i * 8 * 8 * 512, upd_b + (size_t)i * GH,
            ln_g + (size_t)i * GH, ln_b + (size_t)i * GH, hf, hbuf);
    }

    int nbs = GN / 16;  // 6250 score blocks
    score_kernel<<<nbs, 256, 0, stream>>>(hbuf, pk_a1, att_b1, att_w2, att_b2,
                                          scores, red);
    final_max_kernel<<<1, 512, 0, stream>>>(red, nbs, scal, g);
    int nbe = (GN + 255) / 256;  // 391
    expsum_kernel<<<nbe, 256, 0, stream>>>(hf, scores, scal, &scal[1], g);
    head_kernel<<<1, GH, 0, stream>>>(g, &scal[1], hd_w1, hd_b1, hd_w2, hd_b2, out);
}
